// ST_Model_22204980920604
// MI455X (gfx1250) — matrix operand order repaired and checked by reading, not compiled, not run on hardware
//
#include <hip/hip_runtime.h>

typedef __attribute__((ext_vector_type(8)))  _Float16 v8h;
typedef __attribute__((ext_vector_type(16))) _Float16 v16h;
typedef __attribute__((ext_vector_type(8)))  float    v8f;

#define DEV __device__ __forceinline__

DEV float gelu_f(float x) { return 0.5f * x * (1.0f + erff(x * 0.70710678118654752f)); }

DEV v16h cat8(v8h a, v8h b) {
  return __builtin_shufflevector(a, b, 0,1,2,3,4,5,6,7,8,9,10,11,12,13,14,15);
}

// ---------------------------------------------------------------------------
// WMMA GEMM:  C[M x Nact] = A[M x K](f16) * B[Npad x K](f16)^T + bias
// A row-major lda=K, B = weight rows (row n holds W[n][0..K-1]) -> x @ W^T.
// Per-wave 16 x (16*NSUB) tile. flags bit0 = GELU. res: C += res (fp32).
// Fragment layouts per CDNA5 ISA 7.12.2 (wave32):
//   A 16x32 f16 : lane<16 -> M=lane, elems[0..7]=K0..7, [8..15]=K16..23
//                 lane>=16 -> elems[0..7]=K8..15, [8..15]=K24..31
//   B 32x16 f16 : lane l -> N=l&15, same K order as A (elems[0..7]=K8h..8h+7, [8..15]=K16+8h..)
//   C 16x16 f32 : lane l -> N=l&15, elem r -> M=(l>>4)*8+r
// K-loop is explicitly double-buffered in two register sets (64 K-elems per
// iteration): loads for stage k+32 are issued a full compute-phase before
// their consuming WMMAs, so s_wait_loadcnt before each WMMA group refers to
// long-in-flight loads instead of ones issued immediately prior.
// ---------------------------------------------------------------------------
template<int NSUB>
__global__ __launch_bounds__(128)
void k_gemm(const _Float16* __restrict__ A, const _Float16* __restrict__ B,
            const float* __restrict__ bias, const float* res,
            float* outf, _Float16* outh,
            int M, int K, int Nact, int flags)
{
  const int lane = threadIdx.x & 31;
  const int wv   = threadIdx.x >> 5;
  const int mt   = blockIdx.y * 4 + wv;          // wave-uniform
  if (mt * 16 >= M) return;                      // whole wave exits: EXEC stays all-1
  const int nn = lane & 15;
  const int hi = lane >> 4;
  const int n0 = blockIdx.x * (16 * NSUB);

  const _Float16* Ap = A + (size_t)(mt * 16 + nn) * K + hi * 8;
  const _Float16* Bp[NSUB];
#pragma unroll
  for (int j = 0; j < NSUB; j++)
    Bp[j] = B + (size_t)(n0 + j * 16 + nn) * K + hi * 8;

  v8f zero = {};
  v8f acc[NSUB];
#pragma unroll
  for (int j = 0; j < NSUB; j++) acc[j] = zero;

  // two pipeline stages in distinct register sets
  v8h a0c, a1c, b0c[NSUB], b1c[NSUB];
  v8h a0n, a1n, b0n[NSUB], b1n[NSUB];

  // preload stage k=0
  a0c = *(const v8h*)(Ap);
  a1c = *(const v8h*)(Ap + 16);
#pragma unroll
  for (int j = 0; j < NSUB; j++) {
    b0c[j] = *(const v8h*)(Bp[j]);
    b1c[j] = *(const v8h*)(Bp[j] + 16);
  }

  if (K > 32) {
    for (int k0 = 0; k0 < K; k0 += 64) {
      // issue loads for stage k0+32 (overlap with compute of stage k0)
      a0n = *(const v8h*)(Ap + k0 + 32);
      a1n = *(const v8h*)(Ap + k0 + 48);
#pragma unroll
      for (int j = 0; j < NSUB; j++) {
        b0n[j] = *(const v8h*)(Bp[j] + k0 + 32);
        b1n[j] = *(const v8h*)(Bp[j] + k0 + 48);
      }
      // compute stage k0 (loads were issued one phase ago)
      {
        v16h af = cat8(a0c, a1c);
#pragma unroll
        for (int j = 0; j < NSUB; j++)
          acc[j] = __builtin_amdgcn_wmma_f32_16x16x32_f16(false, af, false,
                     cat8(b0c[j], b1c[j]), (short)0, acc[j], false, false);
      }
      // issue loads for stage k0+64 (overlap with compute of stage k0+32)
      if (k0 + 64 < K) {
        a0c = *(const v8h*)(Ap + k0 + 64);
        a1c = *(const v8h*)(Ap + k0 + 80);
#pragma unroll
        for (int j = 0; j < NSUB; j++) {
          b0c[j] = *(const v8h*)(Bp[j] + k0 + 64);
          b1c[j] = *(const v8h*)(Bp[j] + k0 + 80);
        }
      }
      // compute stage k0+32
      {
        v16h af = cat8(a0n, a1n);
#pragma unroll
        for (int j = 0; j < NSUB; j++)
          acc[j] = __builtin_amdgcn_wmma_f32_16x16x32_f16(false, af, false,
                     cat8(b0n[j], b1n[j]), (short)0, acc[j], false, false);
      }
    }
  } else {
    // K == 32: single step
    v16h af = cat8(a0c, a1c);
#pragma unroll
    for (int j = 0; j < NSUB; j++)
      acc[j] = __builtin_amdgcn_wmma_f32_16x16x32_f16(false, af, false,
                 cat8(b0c[j], b1c[j]), (short)0, acc[j], false, false);
  }

#pragma unroll
  for (int j = 0; j < NSUB; j++) {
    int col = n0 + j * 16 + nn;
    if (col < Nact) {
      float bb = bias ? bias[col] : 0.f;
#pragma unroll
      for (int r = 0; r < 8; r++) {
        int row = mt * 16 + hi * 8 + r;
        float v = acc[j][r] + bb;
        if (flags & 1) v = gelu_f(v);
        size_t o = (size_t)row * Nact + col;
        if (res)  v += res[o];
        if (outf) outf[o] = v;
        if (outh) outh[o] = (_Float16)v;
      }
    }
  }
}

// -------------------------- RMSNorm (wave per row) --------------------------
template<int LEN>
__global__ void k_rmsnorm(const float* __restrict__ in, const float* __restrict__ g,
                          float* outf, _Float16* outh, long long rows)
{
  int lane = threadIdx.x & 31;
  long long row = (long long)blockIdx.x * 4 + (threadIdx.x >> 5);
  if (row >= rows) return;
  const float* p = in + (size_t)row * LEN;
  constexpr int PER = LEN / 32;
  float v[PER]; float ss = 0.f;
#pragma unroll
  for (int i = 0; i < PER; i++) { v[i] = p[lane + i * 32]; ss += v[i] * v[i]; }
#pragma unroll
  for (int o = 16; o > 0; o >>= 1) ss += __shfl_xor(ss, o, 32);
  float s = rsqrtf(ss * (1.0f / LEN) + 1e-6f);
#pragma unroll
  for (int i = 0; i < PER; i++) {
    float ov = v[i] * s * g[lane + i * 32];
    if (outf) outf[(size_t)row * LEN + lane + i * 32] = ov;
    if (outh) outh[(size_t)row * LEN + lane + i * 32] = (_Float16)ov;
  }
}

// -------- embed: H[n,t,v,d] = sum_c x[n,c,t,v]*W[d,c] + b[d] + pe[v,d] ------
__global__ void k_embed(const float* __restrict__ x, const float* __restrict__ w,
                        const float* __restrict__ b, const float* __restrict__ pe,
                        float* __restrict__ H, int T, long long total)
{
  size_t idx = (size_t)blockIdx.x * 256 + threadIdx.x;
  if (idx >= (size_t)total) return;
  int d = idx & 127;
  size_t q = idx >> 7;
  int v = (int)(q % 25); q /= 25;
  int t = (int)(q % T);  size_t n = q / T;
  size_t cs = (size_t)T * 25;
  const float* xp = x + (n * 3) * cs + (size_t)t * 25 + v;
  float a = b[d] + pe[v * 128 + d];
  a += xp[0] * w[d * 3 + 0] + xp[cs] * w[d * 3 + 1] + xp[2 * cs] * w[d * 3 + 2];
  H[idx] = a;
}

// -------- token mixing over V=25 (mode 0: gelu->out, mode 1: out += ) ------
__global__ void k_tokenmix(const float* __restrict__ in, const float* __restrict__ w,
                           const float* __restrict__ b, float* out,
                           long long total, int mode)
{
  size_t idx = (size_t)blockIdx.x * 256 + threadIdx.x;
  if (idx >= (size_t)total) return;
  int d = idx & 127;
  size_t q = idx >> 7;
  int j = (int)(q % 25);
  size_t nt = q / 25;
  const float* src = in + nt * 3200 + d;     // stride 128 over joint index
  float a = b[j];
#pragma unroll
  for (int ii = 0; ii < 25; ii++) a += src[(size_t)ii * 128] * w[j * 25 + ii];
  if (mode == 0) out[idx] = gelu_f(a);
  else           out[idx] += a;
}

// --------------- transposes (N,T,V,D) <-> (N*V,T,D), f16 shadow ------------
__global__ void k_t_fwd(const float* __restrict__ H, float* __restrict__ HF,
                        _Float16* __restrict__ HFh, int T, long long total)
{
  size_t idx = (size_t)blockIdx.x * 256 + threadIdx.x;
  if (idx >= (size_t)total) return;
  int d = idx & 127;
  size_t q = idx >> 7;
  int t = (int)(q % T); q /= T;
  int v = (int)(q % 25); size_t n = q / 25;
  float val = H[(((size_t)n * T + t) * 25 + v) * 128 + d];
  HF[idx] = val;
  if (HFh) HFh[idx] = (_Float16)val;
}
__global__ void k_t_bwd(const float* __restrict__ HF, float* __restrict__ H,
                        int T, long long total)
{
  size_t idx = (size_t)blockIdx.x * 256 + threadIdx.x;
  if (idx >= (size_t)total) return;
  int d = idx & 127;
  size_t q = idx >> 7;
  int v = (int)(q % 25); q /= 25;
  int t = (int)(q % T);  size_t n = q / T;
  H[idx] = HF[(((size_t)n * 25 + v) * T + t) * 128 + d];
}

// --------- window gather: pad + roll(-shift) + pick 32-ch half, f16 --------
__global__ void k_gather(const float* __restrict__ XI, _Float16* __restrict__ XW,
                         int Tp, int Tc, int shift, int co, long long total)
{
  size_t idx = (size_t)blockIdx.x * 256 + threadIdx.x;
  if (idx >= (size_t)total) return;
  int c = idx & 31;
  size_t r = idx >> 5;
  int pp = (int)(r % Tp); size_t nv = r / Tp;
  int t = pp + shift; if (t >= Tp) t -= Tp;
  _Float16 val = (_Float16)0.f;
  if (t < Tc) val = (_Float16)XI[(nv * (size_t)Tc + t) * 64 + co + c];
  XW[idx] = val;
}

// --------- window attention core: heads=2, hd=16, bias table + mask --------
template<int WS>
__global__ void k_attn(const float* __restrict__ QKV, const float* __restrict__ bt,
                       _Float16* __restrict__ AO, int Tp, int shift, long long total)
{
  size_t tid = (size_t)blockIdx.x * 256 + threadIdx.x;
  if (tid >= (size_t)total) return;
  int n = (int)(tid % WS);
  int h = (int)((tid / WS) & 1);
  size_t w = tid / (2 * WS);
  size_t rowb = w * WS;
  int pbase = (int)(rowb % (size_t)Tp);
  float q[16];
  const float* Qp = QKV + (rowb + n) * 96 + h * 16;
#pragma unroll
  for (int c = 0; c < 16; c++) q[c] = Qp[c] * 0.25f;     // hd^-0.5 = 1/4
  int segn = 0;
  if (shift > 0) { int pn = pbase + n; segn = (pn >= Tp - WS) ? ((pn >= Tp - shift) ? 2 : 1) : 0; }
  float s[WS]; float mx = -1e30f;
#pragma unroll
  for (int m = 0; m < WS; m++) {
    const float* Kp = QKV + (rowb + m) * 96 + 32 + h * 16;
    float a = 0.f;
#pragma unroll
    for (int c = 0; c < 16; c++) a += q[c] * Kp[c];
    a += bt[(n - m + WS - 1) * 2 + h];
    if (shift > 0) {
      int pm = pbase + m; int segm = (pm >= Tp - WS) ? ((pm >= Tp - shift) ? 2 : 1) : 0;
      if (segm != segn) a -= 100.f;
    }
    s[m] = a; mx = fmaxf(mx, a);
  }
  float sum = 0.f;
#pragma unroll
  for (int m = 0; m < WS; m++) { s[m] = expf(s[m] - mx); sum += s[m]; }
  float inv = 1.f / sum;
  _Float16* Op = AO + (rowb + n) * 32 + h * 16;
#pragma unroll
  for (int c = 0; c < 16; c++) {
    float o = 0.f;
#pragma unroll
    for (int m = 0; m < WS; m++) o += s[m] * QKV[(rowb + m) * 96 + 64 + h * 16 + c];
    Op[c] = (_Float16)(o * inv);
  }
}

// ------------- scatter back: roll(+shift) + crop, into concat f16 ----------
__global__ void k_scatter(const float* __restrict__ PO, _Float16* __restrict__ CC,
                          int Tp, int Tc, int shift, int co, long long total)
{
  size_t idx = (size_t)blockIdx.x * 256 + threadIdx.x;
  if (idx >= (size_t)total) return;
  int c = idx & 31;
  size_t r = idx >> 5;
  int t = (int)(r % Tc); size_t nv = r / Tc;
  int pp = t - shift; if (pp < 0) pp += Tp;
  CC[(nv * (size_t)Tc + t) * 64 + co + c] = (_Float16)PO[(nv * (size_t)Tp + pp) * 32 + c];
}

// ------------- im2col for stride-2 k=3 pad=1 conv over time ----------------
__global__ void k_im2col(const float* __restrict__ HF, _Float16* __restrict__ IC,
                         int Tc, int Tn, long long total)
{
  size_t idx = (size_t)blockIdx.x * 256 + threadIdx.x;
  if (idx >= (size_t)total) return;
  int c = (int)(idx % 384);
  size_t q = idx / 384;
  int t2 = (int)(q % Tn); size_t nv = q / Tn;
  int k = c >> 7, din = c & 127;
  int tin = 2 * t2 + k - 1;
  float v = 0.f;
  if (tin >= 0 && tin < Tc) v = HF[(nv * (size_t)Tc + tin) * 128 + din];
  IC[idx] = (_Float16)v;
}

// ------------------------- mean over joints (V=25) -------------------------
__global__ void k_meanV(const float* __restrict__ H, float* __restrict__ HM,
                        _Float16* __restrict__ HMh, long long total)
{
  size_t idx = (size_t)blockIdx.x * 256 + threadIdx.x;
  if (idx >= (size_t)total) return;
  int d = idx & 127;
  size_t q = idx >> 7;                         // n*64 + t
  const float* p = H + (q * 25) * 128 + d;
  float a = 0.f;
#pragma unroll
  for (int v = 0; v < 25; v++) a += p[(size_t)v * 128];
  a *= (1.f / 25.f);
  HM[idx] = a; HMh[idx] = (_Float16)a;
}

// ----------------- attentive pool (entropy-confidence weights) -------------
__global__ void k_pool(const float* __restrict__ logits, const float* __restrict__ hm,
                       float* __restrict__ pooled)
{
  __shared__ float wsh[64];
  __shared__ float tot;
  int n = blockIdx.x, tid = threadIdx.x;
  if (tid < 64) {
    const float* L = logits + ((size_t)n * 64 + tid) * 60;
    float mx = -1e30f;
    for (int k = 0; k < 60; k++) mx = fmaxf(mx, L[k]);
    float sum = 0.f;
    for (int k = 0; k < 60; k++) sum += expf(L[k] - mx);
    float ent = 0.f;
    for (int k = 0; k < 60; k++) { float p = expf(L[k] - mx) / sum; ent -= p * logf(p + 1e-8f); }
    wsh[tid] = 1.f + 1.f / (1.f + ent);
  }
  __syncthreads();
  if (tid == 0) { float s = 0.f; for (int t = 0; t < 64; t++) s += wsh[t]; tot = s + 1e-8f; }
  __syncthreads();
  if (tid < 128) {
    float a = 0.f;
    for (int t = 0; t < 64; t++) a += wsh[t] * hm[((size_t)n * 64 + t) * 128 + tid];
    pooled[(size_t)n * 128 + tid] = a / tot;
  }
}

// ------------------ weight conversion f32 -> f16, row padded ---------------
__global__ void k_w2h(const float* __restrict__ src, _Float16* __restrict__ dst,
                      int rows, int K, long long total)
{
  size_t idx = (size_t)blockIdx.x * 256 + threadIdx.x;
  if (idx >= (size_t)total) return;
  long long r = (long long)(idx / K);
  dst[idx] = (r < rows) ? (_Float16)src[idx] : (_Float16)0.f;
}
// conv weight (D,D,3) -> (128 x 384) with col = k*128 + din
__global__ void k_cvw(const float* __restrict__ cw, _Float16* __restrict__ dst)
{
  int idx = blockIdx.x * 256 + threadIdx.x;
  if (idx >= 128 * 384) return;
  int dout = idx / 384, c = idx % 384, k = c >> 7, din = c & 127;
  dst[idx] = (_Float16)cw[(dout * 128 + din) * 3 + k];
}

// ---------------------------------------------------------------------------
static inline unsigned nblk(long long n) { return (unsigned)((n + 255) / 256); }

static void launch_gemm(hipStream_t s, const _Float16* A, const _Float16* B,
                        const float* bias, const float* res, float* outf, _Float16* outh,
                        long long M, int K, int Npad, int Nact, int flags)
{
  dim3 blk(128, 1, 1);
  unsigned gy = (unsigned)((M / 16 + 3) / 4);
  if (Npad % 64 == 0) {
    dim3 g(Npad / 64, gy, 1);
    k_gemm<4><<<g, blk, 0, s>>>(A, B, bias, res, outf, outh, (int)M, K, Nact, flags);
  } else {
    dim3 g(Npad / 32, gy, 1);
    k_gemm<2><<<g, blk, 0, s>>>(A, B, bias, res, outf, outh, (int)M, K, Nact, flags);
  }
}

extern "C" void kernel_launch(void* const* d_in, const int* in_sizes, int n_in,
                              void* d_out, int out_size, void* d_ws, size_t ws_size,
                              hipStream_t stream)
{
  (void)in_sizes; (void)n_in; (void)out_size; (void)ws_size;
  const float* x = (const float*)d_in[0];
  int p = 1;
  auto F = [&](int& i) { return (const float*)d_in[i++]; };

  // JAX pytree flatten order: dict keys sorted alphabetically.
  const float *ds_cb[2], *ds_cw[2], *ds_n[2];
  for (int i = 0; i < 2; i++) { ds_cb[i] = F(p); ds_cw[i] = F(p); ds_n[i] = F(p); }
  const float* pe = F(p); const float* eb = F(p); const float* ew = F(p);          // embed: pe, proj.b, proj.w
  const float* fcb = F(p); const float* fcw = F(p); const float* hnorm = F(p);     // head: fc.b, fc.w, norm
  const float* poolb = F(p); const float* poolw = F(p);                            // pool: b, w
  struct SP { const float *cm1b,*cm1w,*cm2b,*cm2w,*n1,*n2,*tm1b,*tm1w,*tm2b,*tm2w; } sp[4];
  for (int i = 0; i < 4; i++) {
    sp[i].cm1b=F(p); sp[i].cm1w=F(p); sp[i].cm2b=F(p); sp[i].cm2w=F(p);
    sp[i].n1=F(p); sp[i].n2=F(p); sp[i].tm1b=F(p); sp[i].tm1w=F(p); sp[i].tm2b=F(p); sp[i].tm2w=F(p);
  }
  struct TB { const float *bt1,*p1b,*p1w,*q1b,*q1w,*bt2,*p2b,*p2w,*q2b,*q2w,
                          *db,*dw,*f1b,*f1w,*f2b,*f2w,*n1,*n2,*ub,*uw; } tb[4];
  for (int i = 0; i < 4; i++) {
    tb[i].bt1=F(p); tb[i].p1b=F(p); tb[i].p1w=F(p); tb[i].q1b=F(p); tb[i].q1w=F(p);
    tb[i].bt2=F(p); tb[i].p2b=F(p); tb[i].p2w=F(p); tb[i].q2b=F(p); tb[i].q2w=F(p);
    tb[i].db=F(p); tb[i].dw=F(p); tb[i].f1b=F(p); tb[i].f1w=F(p); tb[i].f2b=F(p); tb[i].f2w=F(p);
    tb[i].n1=F(p); tb[i].n2=F(p); tb[i].ub=F(p); tb[i].uw=F(p);
  }

  // ----------------------------- workspace -----------------------------
  char* wp = (char*)d_ws;
  auto alloc = [&](size_t bytes) -> void* {
    void* r = (void*)wp; wp += (bytes + 255) & ~(size_t)255; return r;
  };
  const long long NB = 64, Vj = 25, T0 = 256, NV = NB * Vj;
  const size_t maxTok = (size_t)NB * T0 * Vj;                    // 409600

  float*    H    = (float*)alloc(maxTok * 128 * 4);
  float*    HF   = (float*)alloc((size_t)NV * T0 * 128 * 4);
  float*    XN   = (float*)alloc(maxTok * 128 * 4);
  _Float16* XNh  = (_Float16*)alloc(maxTok * 128 * 2);
  float*    TMP  = (float*)alloc(maxTok * 128 * 4);
  _Float16* G1H  = (_Float16*)alloc(maxTok * 256 * 2);
  _Float16* HFh  = (_Float16*)alloc((size_t)NV * T0 * 128 * 2);
  float*    XI0  = (float*)alloc((size_t)NV * T0 * 64 * 4);
  float*    XI   = (float*)alloc((size_t)NV * T0 * 64 * 4);
  _Float16* XW   = (_Float16*)alloc((size_t)NV * 260 * 32 * 2);
  _Float16* AO   = (_Float16*)alloc((size_t)NV * 260 * 32 * 2);
  _Float16* CC   = (_Float16*)alloc((size_t)NV * T0 * 64 * 2);
  float*    HM   = (float*)alloc((size_t)NB * 64 * 128 * 4);
  _Float16* HMh  = (_Float16*)alloc((size_t)NB * 64 * 128 * 2);
  float*    LG   = (float*)alloc((size_t)NB * 64 * 60 * 4);
  float*    PL   = (float*)alloc((size_t)NB * 128 * 4);
  _Float16* PLh  = (_Float16*)alloc((size_t)NB * 128 * 2);
  // liveness-based aliases
  float*    QKV  = (float*)G1H;       // branch phase only; G1H dead then
  float*    PO   = XI0;               // XI0 dead after rmsnorm into XI
  _Float16* IC   = (_Float16*)TMP;    // TMP only live inside spatial block
  float*    DSb  = XN;                // XN only live inside spatial block

  // f16 weight buffers
  _Float16 *Wcm1[4], *Wcm2[4], *Wf1[4], *Wf2[4], *Wdn[4], *Wup[4],
           *Wq1[4], *Wp1[4], *Wq2[4], *Wp2[4], *Wcv[2], *Wpool, *Whead;
  for (int i = 0; i < 4; i++) {
    Wcm1[i] = (_Float16*)alloc(256 * 128 * 2);
    Wcm2[i] = (_Float16*)alloc(128 * 256 * 2);
    Wf1[i]  = (_Float16*)alloc(256 * 128 * 2);
    Wf2[i]  = (_Float16*)alloc(128 * 256 * 2);
    Wdn[i]  = (_Float16*)alloc(64 * 128 * 2);
    Wup[i]  = (_Float16*)alloc(128 * 64 * 2);
    Wq1[i]  = (_Float16*)alloc(96 * 32 * 2);
    Wp1[i]  = (_Float16*)alloc(32 * 32 * 2);
    Wq2[i]  = (_Float16*)alloc(96 * 32 * 2);
    Wp2[i]  = (_Float16*)alloc(32 * 32 * 2);
  }
  Wcv[0] = (_Float16*)alloc(128 * 384 * 2);
  Wcv[1] = (_Float16*)alloc(128 * 384 * 2);
  Wpool  = (_Float16*)alloc(64 * 128 * 2);
  Whead  = (_Float16*)alloc(64 * 128 * 2);

  auto w2h = [&](const float* src, _Float16* dst, int rows, int rowsPad, int K) {
    long long tot = (long long)rowsPad * K;
    k_w2h<<<nblk(tot), 256, 0, stream>>>(src, dst, rows, K, tot);
  };
  for (int i = 0; i < 4; i++) {
    w2h(sp[i].cm1w, Wcm1[i], 256, 256, 128);
    w2h(sp[i].cm2w, Wcm2[i], 128, 128, 256);
    w2h(tb[i].f1w,  Wf1[i],  256, 256, 128);
    w2h(tb[i].f2w,  Wf2[i],  128, 128, 256);
    w2h(tb[i].dw,   Wdn[i],  64,  64,  128);
    w2h(tb[i].uw,   Wup[i],  128, 128, 64);
    w2h(tb[i].q1w,  Wq1[i],  96,  96,  32);
    w2h(tb[i].p1w,  Wp1[i],  32,  32,  32);
    w2h(tb[i].q2w,  Wq2[i],  96,  96,  32);
    w2h(tb[i].p2w,  Wp2[i],  32,  32,  32);
  }
  k_cvw<<<nblk(128 * 384), 256, 0, stream>>>(ds_cw[0], Wcv[0]);
  k_cvw<<<nblk(128 * 384), 256, 0, stream>>>(ds_cw[1], Wcv[1]);
  w2h(poolw, Wpool, 60, 64, 128);
  w2h(fcw,   Whead, 60, 64, 128);

  auto rms128 = [&](const float* in, const float* g, float* of, _Float16* oh, long long rows) {
    k_rmsnorm<128><<<(unsigned)((rows + 3) / 4), 128, 0, stream>>>(in, g, of, oh, rows);
  };
  auto rms64 = [&](const float* in, const float* g, float* of, _Float16* oh, long long rows) {
    k_rmsnorm<64><<<(unsigned)((rows + 3) / 4), 128, 0, stream>>>(in, g, of, oh, rows);
  };

  const int WSv[2] = {10, 20};
  const int SH[4][2] = {{0, 0}, {5, 10}, {0, 0}, {5, 10}};

  // -------------------------------- embed -------------------------------
  long long totH = (long long)maxTok * 128;
  k_embed<<<nblk(totH), 256, 0, stream>>>(x, ew, eb, pe, H, (int)T0, totH);

  // ------------------------------- stages -------------------------------
  int Tc = 256;
  for (int i = 0; i < 4; i++) {
    long long rowsTok = (long long)NB * Tc * Vj;
    long long elems   = rowsTok * 128;

    // spatial block
    rms128(H, sp[i].n1, XN, nullptr, rowsTok);
    k_tokenmix<<<nblk(elems), 256, 0, stream>>>(XN, sp[i].tm1w, sp[i].tm1b, TMP, elems, 0);
    k_tokenmix<<<nblk(elems), 256, 0, stream>>>(TMP, sp[i].tm2w, sp[i].tm2b, H, elems, 1);
    rms128(H, sp[i].n2, nullptr, XNh, rowsTok);
    launch_gemm(stream, XNh, Wcm1[i], sp[i].cm1b, nullptr, nullptr, G1H, rowsTok, 128, 256, 256, 1);
    launch_gemm(stream, G1H, Wcm2[i], sp[i].cm2b, H, H, nullptr, rowsTok, 256, 128, 128, 0);

    // temporal (swin) block on (N*V, Tc, 128)
    long long totF = (long long)NV * Tc * 128;
    k_t_fwd<<<nblk(totF), 256, 0, stream>>>(H, HF, HFh, Tc, totF);
    long long rowsT = (long long)NV * Tc;
    launch_gemm(stream, HFh, Wdn[i], tb[i].db, nullptr, XI0, nullptr, rowsT, 128, 64, 64, 0);
    rms64(XI0, tb[i].n1, XI, nullptr, rowsT);

    for (int b2 = 0; b2 < 2; b2++) {
      int ws = WSv[b2], shift = SH[i][b2], co = b2 * 32;
      int Tp = Tc + ((ws - (Tc % ws)) % ws);
      long long rowsW = (long long)NV * Tp;
      const float*    qb = b2 ? tb[i].q2b : tb[i].q1b;
      const _Float16* qw = b2 ? Wq2[i] : Wq1[i];
      const float*    pb = b2 ? tb[i].p2b : tb[i].p1b;
      const _Float16* pw = b2 ? Wp2[i] : Wp1[i];
      const float*    bt = b2 ? tb[i].bt2 : tb[i].bt1;

      long long gtot = rowsW * 32;
      k_gather<<<nblk(gtot), 256, 0, stream>>>(XI, XW, Tp, Tc, shift, co, gtot);
      launch_gemm(stream, XW, qw, qb, nullptr, QKV, nullptr, rowsW, 32, 96, 96, 0);
      long long atot = (long long)NV * Tp * 2;
      if (ws == 10) k_attn<10><<<nblk(atot), 256, 0, stream>>>(QKV, bt, AO, Tp, shift, atot);
      else          k_attn<20><<<nblk(atot), 256, 0, stream>>>(QKV, bt, AO, Tp, shift, atot);
      launch_gemm(stream, AO, pw, pb, nullptr, PO, nullptr, rowsW, 32, 32, 32, 0);
      long long stot = (long long)NV * Tc * 32;
      k_scatter<<<nblk(stot), 256, 0, stream>>>(PO, CC, Tp, Tc, shift, co, stot);
    }

    launch_gemm(stream, CC, Wup[i], tb[i].ub, HF, HF, nullptr, rowsT, 64, 128, 128, 0);
    rms128(HF, tb[i].n2, nullptr, XNh, rowsT);
    launch_gemm(stream, XNh, Wf1[i], tb[i].f1b, nullptr, nullptr, G1H, rowsT, 128, 256, 256, 1);
    launch_gemm(stream, G1H, Wf2[i], tb[i].f2b, HF, HF, nullptr, rowsT, 256, 128, 128, 0);

    if (i < 2) {
      int Tn = Tc / 2;
      long long rowsD = (long long)NV * Tn;
      long long ictot = rowsD * 384;
      k_im2col<<<nblk(ictot), 256, 0, stream>>>(HF, IC, Tc, Tn, ictot);
      launch_gemm(stream, IC, Wcv[i], ds_cb[i], nullptr, DSb, nullptr, rowsD, 384, 128, 128, 0);
      rms128(DSb, ds_n[i], HF, nullptr, rowsD);
      Tc = Tn;
    }
    long long totB = (long long)NB * Tc * Vj * 128;
    k_t_bwd<<<nblk(totB), 256, 0, stream>>>(HF, H, Tc, totB);
  }

  // ------------------------ pooling + head (Tc == 64) --------------------
  long long mtot = (long long)NB * 64 * 128;
  k_meanV<<<nblk(mtot), 256, 0, stream>>>(H, HM, HMh, mtot);
  launch_gemm(stream, HMh, Wpool, poolb, nullptr, LG, nullptr, (long long)NB * 64, 128, 64, 60, 0);
  k_pool<<<dim3(64), dim3(128), 0, stream>>>(LG, HM, PL);
  rms128(PL, hnorm, nullptr, PLh, NB);
  launch_gemm(stream, PLh, Whead, fcb, nullptr, (float*)d_out, nullptr, NB, 128, 64, 60, 0);
}
